// SelfAttention_29033978921186
// MI455X (gfx1250) — compile-verified, matrix operand order checked by reading, not run on hardware
//
#include <hip/hip_runtime.h>
#include <hip/hip_bf16.h>

// ---------------------------------------------------------------------------
// Self-attention, bf16 WMMA pipeline for gfx1250 (MI455X)
//   x:[2,2048,1024] f32, Wq/Wk/Wv/Wu:[1024,1024] f32 (y = x @ W^T), bu:[1024]
// All GEMMs feed V_WMMA_F32_16X16X32_BF16 from async-staged LDS slabs.
// ---------------------------------------------------------------------------

typedef __attribute__((ext_vector_type(16))) __bf16         bf16x16;
typedef __attribute__((ext_vector_type(8)))  float          f32x8;
typedef __attribute__((ext_vector_type(8)))  unsigned short u16x8;
typedef int v4i __attribute__((vector_size(16)));   // matches builtin param type

#define TT    2048
#define KDIM  1024
#define NHEAD 16
#define HDIM  64

#define AS1 __attribute__((address_space(1)))
#define AS3 __attribute__((address_space(3)))

#if defined(__has_builtin)
#  if __has_builtin(__builtin_amdgcn_global_load_async_to_lds_b128)
#    define USE_ASYNC_LDS 1
#  endif
#  if __has_builtin(__builtin_amdgcn_s_wait_asynccnt)
#    define USE_WAIT_ASYNC_BUILTIN 1
#  endif
#endif

__device__ __forceinline__ unsigned short f2bf(float f) {
    unsigned u = __float_as_uint(f);
    unsigned r = u + 0x7FFFu + ((u >> 16) & 1u);   // round-to-nearest-even
    return (unsigned short)(r >> 16);
}

__device__ __forceinline__ f32x8 wmma_bf16(bf16x16 a, bf16x16 b, f32x8 c) {
    // (neg_a, A, neg_b, B, c_mod, C, reuse_a, reuse_b)
    return __builtin_amdgcn_wmma_f32_16x16x32_bf16(false, a, false, b,
                                                   (short)0, c, false, false);
}

// 16-byte global -> LDS copy. Async (GLOBAL_LOAD_ASYNC_TO_LDS_B128, tracked by
// ASYNCcnt) when the builtin exists; plain register copy otherwise.
__device__ __forceinline__ void lds_copy_b128(unsigned short* dst,
                                              const unsigned short* src) {
#ifdef USE_ASYNC_LDS
    unsigned long long g = (unsigned long long)src;            // AS1 == flat-global
    unsigned           l = (unsigned)(unsigned long long)dst;  // low 32b = LDS offset
    __builtin_amdgcn_global_load_async_to_lds_b128((AS1 v4i*)g, (AS3 v4i*)l, 0, 0);
#else
    *(u16x8*)dst = *(const u16x8*)src;
#endif
}

__device__ __forceinline__ void wait_async_lds() {
#ifdef USE_ASYNC_LDS
#  ifdef USE_WAIT_ASYNC_BUILTIN
    __builtin_amdgcn_s_wait_asynccnt(0);
#  else
    asm volatile("s_wait_asynccnt 0x0" ::: "memory");
#  endif
#endif
}

// Load one 16x32 bf16 A/B fragment for this lane.
// CDNA5 16-bit layout: element e -> K = (e&7) + (e>=8?16:0) + khalf,
// khalf = (lane>=16 ? 8 : 0). Two contiguous 8-element runs -> two b128 loads.
__device__ __forceinline__ bf16x16 load_frag(const unsigned short* rowbase, int khalf) {
    union { u16x8 p[2]; bf16x16 v; } u;
    u.p[0] = *(const u16x8*)(rowbase + khalf);
    u.p[1] = *(const u16x8*)(rowbase + 16 + khalf);
    return u.v;
}

// ---------------------------------------------------------------------------
// f32 -> bf16 conversion
// ---------------------------------------------------------------------------
__global__ void cvt_bf16_kernel(const float* __restrict__ src,
                                unsigned short* __restrict__ dst, int n) {
    int i = blockIdx.x * blockDim.x + threadIdx.x;
    if (i < n) dst[i] = f2bf(src[i]);
}

// ---------------------------------------------------------------------------
// GEMM: out[M,N] = scale * (A[M,K] @ W[N,K]^T) (+ bias), bf16 in, bf16/f32 out
// Block: 256 threads = 8 waves, 128(M) x 64(N) tile; wave = 32x32 via 2x2 WMMA.
// A/B slabs for each 32-wide K step are double-buffered in LDS via async copies
// so step k+1's GLOBAL_LOAD_ASYNC_TO_LDS overlaps step k's WMMAs.
// ---------------------------------------------------------------------------
#define ASTR 40   // padded LDS row stride (halves): 80B, 16B aligned, de-banked

__global__ __launch_bounds__(256) void gemm_bf16_kernel(
    const unsigned short* __restrict__ A, const unsigned short* __restrict__ W,
    unsigned short* __restrict__ outb, float* __restrict__ outf,
    const float* __restrict__ bias, int M, int Kd, int N, float scale)
{
    __shared__ unsigned short Asl[2][128 * ASTR];
    __shared__ unsigned short Bsl[2][64 * ASTR];

    const int tid   = threadIdx.x;
    const int lane  = tid & 31;
    const int wv    = tid >> 5;
    const int wm    = wv & 3;
    const int wn    = wv >> 2;
    const int mr    = lane & 15;
    const int khalf = (lane >> 4) << 3;
    const int mblk  = blockIdx.y * 128;
    const int nblk  = blockIdx.x * 64;

    // stage one 32-wide K slab (A:128x32, B:64x32) into LDS buffer `buf`
    auto stage = [&](int buf, int k0) {
        #pragma unroll
        for (int v = 0; v < 2; ++v) {                 // A: 512 b128 vectors
            const int vec = tid + (v << 8);
            const int row = vec >> 2, ko = (vec & 3) << 3;
            lds_copy_b128(&Asl[buf][row * ASTR + ko],
                          A + (size_t)(mblk + row) * Kd + k0 + ko);
        }
        {                                             // B: 256 b128 vectors
            const int row = tid >> 2, ko = (tid & 3) << 3;
            lds_copy_b128(&Bsl[buf][row * ASTR + ko],
                          W + (size_t)(nblk + row) * Kd + k0 + ko);
        }
    };

    f32x8 acc00 = {}, acc01 = {}, acc10 = {}, acc11 = {};

    stage(0, 0);
    wait_async_lds();
    __syncthreads();

    for (int k0 = 0; k0 < Kd; k0 += 32) {
        const int cur = (k0 >> 5) & 1;
        if (k0 + 32 < Kd) stage(cur ^ 1, k0 + 32);    // async prefetch next slab

        const unsigned short* ab = &Asl[cur][0];
        const unsigned short* bb = &Bsl[cur][0];
        bf16x16 a0 = load_frag(ab + (wm * 32 + mr)      * ASTR, khalf);
        bf16x16 a1 = load_frag(ab + (wm * 32 + 16 + mr) * ASTR, khalf);
        bf16x16 b0 = load_frag(bb + (wn * 32 + mr)      * ASTR, khalf);
        bf16x16 b1 = load_frag(bb + (wn * 32 + 16 + mr) * ASTR, khalf);
        acc00 = wmma_bf16(a0, b0, acc00);
        acc01 = wmma_bf16(a0, b1, acc01);
        acc10 = wmma_bf16(a1, b0, acc10);
        acc11 = wmma_bf16(a1, b1, acc11);

        wait_async_lds();                             // own copies for next slab
        __syncthreads();                              // everyone's copies visible
    }

    const int rbase = (lane >> 4) << 3;               // C/D layout: rows 0-7 | 8-15
    f32x8 accs[2][2] = { { acc00, acc01 }, { acc10, acc11 } };
    #pragma unroll
    for (int i = 0; i < 2; ++i) {
        #pragma unroll
        for (int j = 0; j < 2; ++j) {
            const int col = nblk + wn * 32 + j * 16 + mr;
            const float bv = bias ? bias[col] : 0.0f;
            #pragma unroll
            for (int r = 0; r < 8; ++r) {
                const int row = mblk + wm * 32 + i * 16 + rbase + r;
                const float v = accs[i][j][r] * scale + bv;
                if (outf) outf[(size_t)row * N + col] = v;
                else      outb[(size_t)row * N + col] = f2bf(v);
            }
        }
    }
}

// ---------------------------------------------------------------------------
// Flash attention: softmax(Q K^T) V per (batch, head).
// Block: 128 threads = 4 waves; each wave owns 16 query rows; 32 keys / iter.
// K tile (shared by all 4 waves) is async-staged row-major into LDS;
// V tile is staged transposed (register scatter) for contiguous PV B-frags.
// ---------------------------------------------------------------------------
#define KTSTR 72   // kt row stride (halves): 144B, 16B aligned
#define VTSTR 40   // vt/pl row stride (halves): 80B

__global__ __launch_bounds__(128) void attn_kernel(
    const unsigned short* __restrict__ Q, const unsigned short* __restrict__ Kb,
    const unsigned short* __restrict__ V, unsigned short* __restrict__ O)
{
    __shared__ unsigned short kt[32 * KTSTR];        // K tile row-major [s][d]
    __shared__ unsigned short vt[HDIM * VTSTR];      // V tile transposed [d][s]
    __shared__ unsigned short pl[4][16 * VTSTR];     // per-wave P transpose slab

    const int lane  = threadIdx.x & 31;
    const int wv    = threadIdx.x >> 5;
    const int b     = blockIdx.z;
    const int h     = blockIdx.y;
    const int q0    = blockIdx.x * 64 + wv * 16;
    const int mr    = lane & 15;
    const int khalf = (lane >> 4) << 3;
    const int rbase = khalf;                         // row half of C/D layout

    const size_t base = (size_t)b * TT * KDIM + (size_t)h * HDIM;

    // Q fragments (16 rows x 64 d) pinned in registers
    const unsigned short* qrow = Q + base + (size_t)(q0 + mr) * KDIM;
    bf16x16 qf0 = load_frag(qrow + 0,  khalf);
    bf16x16 qf1 = load_frag(qrow + 32, khalf);

    float mrow[8], lrow[8];
    f32x8 acc[4] = {};
    #pragma unroll
    for (int r = 0; r < 8; ++r) { mrow[r] = -3.0e38f; lrow[r] = 0.0f; }

    for (int sb = 0; sb < TT; sb += 32) {
        __syncthreads();                             // previous tile fully read
        // K tile: 256 b128 vectors, async global -> LDS (shared by all waves)
        for (int vec = threadIdx.x; vec < 256; vec += 128) {
            const int s = vec >> 3, ko = (vec & 7) << 3;
            lds_copy_b128(kt + s * KTSTR + ko,
                          Kb + base + (size_t)(sb + s) * KDIM + ko);
        }
        // V tile: load row-major, scatter transposed into vt[d][s]
        for (int vec = threadIdx.x; vec < 256; vec += 128) {
            const int s = vec >> 3, dblk = (vec & 7) << 3;
            u16x8 d8 = *(const u16x8*)(V + base + (size_t)(sb + s) * KDIM + dblk);
            #pragma unroll
            for (int e = 0; e < 8; ++e) vt[(dblk + e) * VTSTR + s] = d8[e];
        }
        wait_async_lds();
        __syncthreads();

        // S = Q K^T : two 16x16 tiles, K-dim = 64 via two WMMAs each
        f32x8 s0 = {}, s1 = {};
        {
            const unsigned short* k0p = kt + (size_t)mr        * KTSTR;
            const unsigned short* k1p = kt + (size_t)(16 + mr) * KTSTR;
            bf16x16 kf;
            kf = load_frag(k0p + 0,  khalf); s0 = wmma_bf16(qf0, kf, s0);
            kf = load_frag(k0p + 32, khalf); s0 = wmma_bf16(qf1, kf, s0);
            kf = load_frag(k1p + 0,  khalf); s1 = wmma_bf16(qf0, kf, s1);
            kf = load_frag(k1p + 32, khalf); s1 = wmma_bf16(qf1, kf, s1);
        }

        // online softmax: rows 0-7 live in lanes 0-15, rows 8-15 in 16-31,
        // so width-16 xor-shuffles reduce across the 16 columns per row.
        float corr[8];
        #pragma unroll
        for (int r = 0; r < 8; ++r) {
            float mx = fmaxf(s0[r], s1[r]);
            #pragma unroll
            for (int off = 1; off < 16; off <<= 1)
                mx = fmaxf(mx, __shfl_xor(mx, off, 16));
            const float mn = fmaxf(mrow[r], mx);
            corr[r] = __expf(mrow[r] - mn);
            mrow[r] = mn;
            const float p0 = __expf(s0[r] - mn);
            const float p1 = __expf(s1[r] - mn);
            s0[r] = p0; s1[r] = p1;
            float ps = p0 + p1;
            #pragma unroll
            for (int off = 1; off < 16; off <<= 1)
                ps += __shfl_xor(ps, off, 16);
            lrow[r] = lrow[r] * corr[r] + ps;
        }
        #pragma unroll
        for (int jd = 0; jd < 4; ++jd)
            #pragma unroll
            for (int r = 0; r < 8; ++r)
                acc[jd][r] = acc[jd][r] * corr[r];

        // transpose P through per-wave LDS: C-layout (lane=col) -> A-layout
        unsigned short* pw = &pl[wv][0];
        #pragma unroll
        for (int r = 0; r < 8; ++r) {
            const int row = rbase + r;
            pw[row * VTSTR + mr]      = f2bf(s0[r]);
            pw[row * VTSTR + 16 + mr] = f2bf(s1[r]);
        }
        bf16x16 pa = load_frag(pw + mr * VTSTR, khalf);  // in-wave DS in-order

        // O += P V : 4 output d-tiles, K-dim(s)=32 -> one WMMA each
        #pragma unroll
        for (int jd = 0; jd < 4; ++jd) {
            bf16x16 vb = load_frag(vt + (jd * 16 + mr) * VTSTR, khalf);
            acc[jd] = wmma_bf16(pa, vb, acc[jd]);
        }
    }

    // normalize and store bf16 head slice
    #pragma unroll
    for (int jd = 0; jd < 4; ++jd) {
        #pragma unroll
        for (int r = 0; r < 8; ++r) {
            const float v = acc[jd][r] / lrow[r];
            const size_t row = (size_t)b * TT + q0 + rbase + r;
            O[row * KDIM + (size_t)h * HDIM + jd * 16 + mr] = f2bf(v);
        }
    }
}

// ---------------------------------------------------------------------------
// launch
// ---------------------------------------------------------------------------
extern "C" void kernel_launch(void* const* d_in, const int* in_sizes, int n_in,
                              void* d_out, int out_size, void* d_ws, size_t ws_size,
                              hipStream_t stream)
{
    const float* x  = (const float*)d_in[0];
    const float* Wq = (const float*)d_in[1];
    const float* Wk = (const float*)d_in[2];
    const float* Wv = (const float*)d_in[3];
    const float* Wu = (const float*)d_in[4];
    const float* bu = (const float*)d_in[5];
    float* out = (float*)d_out;

    const size_t NX = (size_t)2 * TT * KDIM;   // 4,194,304
    const size_t NW = (size_t)KDIM * KDIM;     // 1,048,576

    unsigned short* ws  = (unsigned short*)d_ws;  // ~48 MiB of scratch used
    unsigned short* xb  = ws;
    unsigned short* wqb = xb  + NX;
    unsigned short* wkb = wqb + NW;
    unsigned short* wvb = wkb + NW;
    unsigned short* wub = wvb + NW;
    unsigned short* Qb  = wub + NW;
    unsigned short* Kbf = Qb  + NX;
    unsigned short* Vbf = Kbf + NX;
    unsigned short* Ob  = Vbf + NX;

    cvt_bf16_kernel<<<(unsigned)((NX + 255) / 256), 256, 0, stream>>>(x,  xb,  (int)NX);
    cvt_bf16_kernel<<<(unsigned)((NW + 255) / 256), 256, 0, stream>>>(Wq, wqb, (int)NW);
    cvt_bf16_kernel<<<(unsigned)((NW + 255) / 256), 256, 0, stream>>>(Wk, wkb, (int)NW);
    cvt_bf16_kernel<<<(unsigned)((NW + 255) / 256), 256, 0, stream>>>(Wv, wvb, (int)NW);
    cvt_bf16_kernel<<<(unsigned)((NW + 255) / 256), 256, 0, stream>>>(Wu, wub, (int)NW);

    const int M = 2 * TT;                       // batch rows flattened
    dim3 gg(KDIM / 64, M / 128);                // (16, 32)
    const float invs = 0.17677669529663688f;    // 1 / 1024^(1/4)

    gemm_bf16_kernel<<<gg, 256, 0, stream>>>(xb, wqb, Qb,  nullptr, nullptr, M, KDIM, KDIM, invs);
    gemm_bf16_kernel<<<gg, 256, 0, stream>>>(xb, wkb, Kbf, nullptr, nullptr, M, KDIM, KDIM, invs);
    gemm_bf16_kernel<<<gg, 256, 0, stream>>>(xb, wvb, Vbf, nullptr, nullptr, M, KDIM, KDIM, 1.0f);

    attn_kernel<<<dim3(TT / 64, NHEAD, 2), 128, 0, stream>>>(Qb, Kbf, Vbf, Ob);

    gemm_bf16_kernel<<<gg, 256, 0, stream>>>(Ob, wub, nullptr, out, bu, M, KDIM, KDIM, 1.0f);
}
